// ClassifierB_53034256171381
// MI455X (gfx1250) — matrix operand order repaired and checked by reading, not compiled, not run on hardware
//
#include <hip/hip_runtime.h>
#include <hip/hip_bf16.h>
#include <math.h>

// ---------------------------------------------------------------------------
// Types for CDNA5 WMMA (wave32): v_wmma_f32_16x16x32_bf16
// ---------------------------------------------------------------------------
typedef __attribute__((ext_vector_type(16))) __bf16 v16bf;
typedef __attribute__((ext_vector_type(8)))  float  v8f;

#define LDSPAD 40   // 32 K-halves + 8 pad halves -> 80B row stride (16B aligned, bank-spread)

// Pure 32-bit integer bf16 round-to-nearest-even: result in TOP 16 bits.
__device__ __forceinline__ unsigned bf16_hi(float f) {
  unsigned u = __float_as_uint(f);
  return u + 0x7FFFu + ((u >> 16) & 1u);
}
// Pack two floats into one dword of bf16 (lo -> [15:0], hi -> [31:16]).
__device__ __forceinline__ unsigned pack_bf16(float lo, float hi) {
  return (bf16_hi(lo) >> 16) | (bf16_hi(hi) & 0xFFFF0000u);
}
__device__ __forceinline__ __bf16 f2bf(float f) {
  unsigned short s = (unsigned short)(bf16_hi(f) >> 16);
  __bf16 b;
  __builtin_memcpy(&b, &s, sizeof(s));
  return b;
}

union FragBF {
  uint4 q[2];
  v16bf v;
};

// Async global->LDS 16B copy (CDNA5 GLOBAL_LOAD_ASYNC_TO_LDS_B128, ASYNCcnt).
__device__ __forceinline__ void async_copy_b128(unsigned lds_byte_addr,
                                                const void* gaddr) {
  unsigned long long ga = (unsigned long long)(size_t)gaddr;
  asm volatile("global_load_async_to_lds_b128 %0, %1, off"
               :: "v"(lds_byte_addr), "v"(ga) : "memory");
}
__device__ __forceinline__ void wait_async0() {
  asm volatile("s_wait_asynccnt 0x0" ::: "memory");
}
// Low 32 bits of a generic LDS pointer == wave-relative LDS byte address.
__device__ __forceinline__ unsigned lds_addr32(const void* p) {
  return (unsigned)(size_t)p;
}

// ---------------------------------------------------------------------------
// WMMA GEMM: C[M,N] = epi(A[M,K] * B[N,K]^T + bias[N])
//   A: fp32 (AF32=true, converted to bf16 while staging) or bf16 row-major
//   B: bf16 row-major [N,K] (weight matrix, dot along K)
//   EPI==0 -> C fp32 = acc + bias
//   EPI==1 -> C bf16 = bf16(tanh(acc + bias))
// Block: 256 threads (8 waves), tile 128(M) x 64(N), K step 32.
// Waves 4x2; each wave -> 32x32 via 2x2 v_wmma_f32_16x16x32_bf16 tiles.
// bf16 tiles staged with GLOBAL_LOAD_ASYNC_TO_LDS_B128 (ASYNCcnt).
// ---------------------------------------------------------------------------
template <int EPI, bool AF32>
__global__ __launch_bounds__(256) void wmma_gemm_kernel(
    const void* __restrict__ Aptr, const __bf16* __restrict__ Bw,
    const float* __restrict__ bias, void* __restrict__ Cptr,
    int M, int N, int K) {
  __shared__ __bf16 As[128 * LDSPAD];
  __shared__ __bf16 Bs[64 * LDSPAD];

  const int tid   = threadIdx.x;
  const int lane  = tid & 31;
  const int wave  = tid >> 5;
  const int waveM = wave >> 1;          // 0..3
  const int waveN = wave & 1;           // 0..1
  const int m0    = blockIdx.y * 128;
  const int n0    = blockIdx.x * 64;

  const int lr = lane & 15;             // row/col within 16
  const int lh = lane >> 4;             // lane half selector

  v8f acc[2][2] = {{{}, {}}, {{}, {}}};

  // staging split: A: 2 threads/row (16 halves each); B: 4 threads/row (8 halves each)
  const int arow = tid >> 1;            // 0..127
  const int acol = (tid & 1) << 4;      // 0 or 16
  const int brow = tid >> 2;            // 0..63
  const int bcol = (tid & 3) << 3;      // 0,8,16,24

  const unsigned lds_a0 = lds_addr32(&As[arow * LDSPAD + acol]);
  const unsigned lds_a1 = lds_a0 + 16;
  const unsigned lds_b  = lds_addr32(&Bs[brow * LDSPAD + bcol]);

  for (int kb = 0; kb < K; kb += 32) {
    __syncthreads();
    // ---- stage B tile [64 x 32] via async DMA into LDS ----
    {
      const __bf16* Bp = Bw + (size_t)(n0 + brow) * K + kb + bcol;
      async_copy_b128(lds_b, Bp);
      if (kb + 32 < K) __builtin_prefetch(Bp + 32, 0, 1);
    }
    // ---- stage A tile [128 x 32] into LDS ----
    if constexpr (AF32) {
      // fp32 source: load, convert to bf16, ds_store (fallback path)
      const float* Ap = (const float*)Aptr + (size_t)(m0 + arow) * K + kb + acol;
      float4 f0 = *(const float4*)(Ap + 0);
      float4 f1 = *(const float4*)(Ap + 4);
      float4 f2 = *(const float4*)(Ap + 8);
      float4 f3 = *(const float4*)(Ap + 12);
      uint4 p0, p1;
      p0.x = pack_bf16(f0.x, f0.y);
      p0.y = pack_bf16(f0.z, f0.w);
      p0.z = pack_bf16(f1.x, f1.y);
      p0.w = pack_bf16(f1.z, f1.w);
      p1.x = pack_bf16(f2.x, f2.y);
      p1.y = pack_bf16(f2.z, f2.w);
      p1.z = pack_bf16(f3.x, f3.y);
      p1.w = pack_bf16(f3.z, f3.w);
      *(uint4*)&As[arow * LDSPAD + acol]     = p0;
      *(uint4*)&As[arow * LDSPAD + acol + 8] = p1;
      if (kb + 32 < K) __builtin_prefetch(Ap + 32, 0, 1);
    } else {
      // bf16 source: straight async DMA copy
      const __bf16* Ap = (const __bf16*)Aptr + (size_t)(m0 + arow) * K + kb + acol;
      async_copy_b128(lds_a0, Ap);
      async_copy_b128(lds_a1, Ap + 8);
      if (kb + 32 < K) __builtin_prefetch(Ap + 32, 0, 1);
    }
    wait_async0();
    __syncthreads();

    // ---- load fragments per documented CDNA5 layouts ----
    FragBF afr[2], bfr[2];
#pragma unroll
    for (int mi = 0; mi < 2; ++mi) {
      // A 16x32 bf16: lane L holds row L&15; halves {k0..k0+7} then {k0+16..k0+23}, k0 = 8*(L>>4)
      const __bf16* p = &As[(waveM * 32 + mi * 16 + lr) * LDSPAD + lh * 8];
      afr[mi].q[0] = *(const uint4*)(p);
      afr[mi].q[1] = *(const uint4*)(p + 16);
    }
#pragma unroll
    for (int ni = 0; ni < 2; ++ni) {
      // B 32x16 bf16: lane L holds column L&15 (= weight row); halves {k0..k0+7} then {k0+16..k0+23}, k0 = 8*(L>>4)
      const __bf16* p = &Bs[(waveN * 32 + ni * 16 + lr) * LDSPAD + lh * 8];
      bfr[ni].q[0] = *(const uint4*)(p);
      bfr[ni].q[1] = *(const uint4*)(p + 16);
    }
#pragma unroll
    for (int mi = 0; mi < 2; ++mi)
#pragma unroll
      for (int ni = 0; ni < 2; ++ni)
        acc[mi][ni] = __builtin_amdgcn_wmma_f32_16x16x32_bf16(
            false, afr[mi].v, false, bfr[ni].v, (short)0, acc[mi][ni],
            false, false);
  }

  // ---- epilogue: C/D layout -> M = base + r + 8*(lane>>4), N = base + (lane&15) ----
#pragma unroll
  for (int mi = 0; mi < 2; ++mi) {
#pragma unroll
    for (int ni = 0; ni < 2; ++ni) {
      const int mb = m0 + waveM * 32 + mi * 16 + lh * 8;
      const int n  = n0 + waveN * 32 + ni * 16 + lr;
      const float bv = bias ? bias[n] : 0.0f;
#pragma unroll
      for (int r = 0; r < 8; ++r) {
        float v = acc[mi][ni][r] + bv;
        if constexpr (EPI == 1) {
          ((__bf16*)Cptr)[(size_t)(mb + r) * N + n] = f2bf(tanhf(v));
        } else {
          ((float*)Cptr)[(size_t)(mb + r) * N + n] = v;
        }
      }
    }
  }
}

// ---------------------------------------------------------------------------
// fp32 -> bf16 conversions (scalar + vectorized-by-4)
// ---------------------------------------------------------------------------
__global__ void cvt_bf16_kernel(const float* __restrict__ in,
                                __bf16* __restrict__ out, int n) {
  int i = blockIdx.x * 256 + threadIdx.x;
  if (i < n) out[i] = f2bf(in[i]);
}

__global__ void cvt4_bf16_kernel(const float* __restrict__ in,
                                 uint2* __restrict__ out, int n4) {
  int i = blockIdx.x * 256 + threadIdx.x;
  if (i < n4) {
    float4 f = ((const float4*)in)[i];
    uint2 o;
    o.x = pack_bf16(f.x, f.y);
    o.y = pack_bf16(f.z, f.w);
    out[i] = o;
  }
}

// zero-init GRU hidden state (fp32 + bf16 mirrors)
__global__ void gru_init_kernel(float* __restrict__ h32,
                                __bf16* __restrict__ hbf, int n) {
  int i = blockIdx.x * 256 + threadIdx.x;
  if (i < n) { h32[i] = 0.0f; hbf[i] = f2bf(0.0f); }
}

// ---------------------------------------------------------------------------
// GRU gate pointwise for timestep t.
//   m in [0,384): m = b*12 + p ; gx row = ((b*40 + t)*12 + p)
// ---------------------------------------------------------------------------
__global__ __launch_bounds__(256) void gru_pointwise_kernel(
    const float* __restrict__ gx, const float* __restrict__ gh,
    float* __restrict__ h32, __bf16* __restrict__ hbf,
    float* __restrict__ hiddens, int t) {
  int idx = blockIdx.x * 256 + threadIdx.x;        // [0, 384*1024)
  int m  = idx >> 10;
  int hh = idx & 1023;
  int b = m / 12, p = m % 12;
  size_t gxrow = (size_t)(b * 40 + t) * 12 + p;
  const float* gxp = gx + gxrow * 3072;
  const float* ghp = gh + (size_t)m * 3072;
  float xr = gxp[hh], xz = gxp[1024 + hh], xn = gxp[2048 + hh];
  float hr = ghp[hh], hz = ghp[1024 + hh], hn = ghp[2048 + hh];
  float r = 1.0f / (1.0f + expf(-(xr + hr)));
  float z = 1.0f / (1.0f + expf(-(xz + hz)));
  float nn = tanhf(xn + r * hn);
  float ho = h32[idx];
  float hnew = (1.0f - z) * nn + z * ho;
  h32[idx] = hnew;
  hbf[idx] = f2bf(hnew);
  hiddens[gxrow * 1024 + hh] = hnew;               // [B,T,P,H]
}

// ---------------------------------------------------------------------------
// Heads: one wave per output row, shfl_xor tree reductions.
// ---------------------------------------------------------------------------
__global__ void action_head_kernel(const float* __restrict__ hiddens,
                                   const float* __restrict__ W,
                                   const float* __restrict__ bias,
                                   float* __restrict__ out) {
  int m = blockIdx.x;                 // 0..15359
  int lane = threadIdx.x;             // 0..31
  const float* hp = hiddens + (size_t)m * 1024;
  float hv[32];
#pragma unroll
  for (int i = 0; i < 32; ++i) hv[i] = hp[lane + i * 32];
#pragma unroll
  for (int a = 0; a < 9; ++a) {
    const float* wp = W + a * 1024;
    float s = 0.0f;
#pragma unroll
    for (int i = 0; i < 32; ++i) s += hv[i] * wp[lane + i * 32];
#pragma unroll
    for (int off = 16; off > 0; off >>= 1) s += __shfl_xor(s, off, 32);
    if (lane == 0) out[m * 9 + a] = s + bias[a];
  }
}

__global__ void activity_head_kernel(const float* __restrict__ hiddens,
                                     const float* __restrict__ W,
                                     const float* __restrict__ bias,
                                     float* __restrict__ out) {
  int bt = blockIdx.x;                // 0..1279 (= b*40 + t)
  int lane = threadIdx.x;
  const float* hp = hiddens + (size_t)bt * 12 * 1024;
  float pv[32];
#pragma unroll
  for (int i = 0; i < 32; ++i) pv[i] = -3.402823466e38f;
  for (int p = 0; p < 12; ++p) {
#pragma unroll
    for (int i = 0; i < 32; ++i)
      pv[i] = fmaxf(pv[i], hp[p * 1024 + lane + i * 32]);
  }
#pragma unroll
  for (int a = 0; a < 8; ++a) {
    const float* wp = W + a * 1024;
    float s = 0.0f;
#pragma unroll
    for (int i = 0; i < 32; ++i) s += pv[i] * wp[lane + i * 32];
#pragma unroll
    for (int off = 16; off > 0; off >>= 1) s += __shfl_xor(s, off, 32);
    if (lane == 0) out[bt * 8 + a] = s + bias[a];
  }
}

// ---------------------------------------------------------------------------
// Host-side launch
// ---------------------------------------------------------------------------
extern "C" void kernel_launch(void* const* d_in, const int* in_sizes, int n_in,
                              void* d_out, int out_size, void* d_ws, size_t ws_size,
                              hipStream_t stream) {
  (void)in_sizes; (void)n_in; (void)out_size;
  const float* feature = (const float*)d_in[0];   // [32,40,12,2048]
  const float* W_embed = (const float*)d_in[1];   // [1024,2048]
  const float* b_embed = (const float*)d_in[2];   // [1024]
  const float* W_ih    = (const float*)d_in[3];   // [3072,1024]
  const float* W_hh    = (const float*)d_in[4];   // [3072,1024]
  const float* b_ih    = (const float*)d_in[5];   // [3072]
  const float* b_hh    = (const float*)d_in[6];   // [3072]
  const float* W_act   = (const float*)d_in[7];   // [9,1024]
  const float* b_act   = (const float*)d_in[8];   // [9]
  const float* W_activ = (const float*)d_in[9];   // [8,1024]
  const float* b_activ = (const float*)d_in[10];  // [8]

  constexpr int B = 32, T = 40, P = 12, E = 2048, H = 1024;
  constexpr int M  = B * T * P;   // 15360
  constexpr int MR = B * P;       // 384

  char* ws = (char*)d_ws;
  size_t off = 0;
  auto carve = [&](size_t bytes) -> char* {
    char* p = ws + off;
    off = (off + bytes + 255) & ~(size_t)255;
    return p;
  };
  __bf16* wembed_bf = (__bf16*)carve((size_t)H * E * 2);
  __bf16* wih_bf    = (__bf16*)carve((size_t)3 * H * H * 2);
  __bf16* whh_bf    = (__bf16*)carve((size_t)3 * H * H * 2);
  __bf16* x_bf      = (__bf16*)carve((size_t)M * H * 2);
  float*  gx        = (float*)carve((size_t)M * 3 * H * 4);
  float*  gh        = (float*)carve((size_t)MR * 3 * H * 4);
  float*  h32       = (float*)carve((size_t)MR * H * 4);
  __bf16* hbf       = (__bf16*)carve((size_t)MR * H * 2);
  float*  hiddens   = (float*)carve((size_t)M * H * 4);
  // Optional: pre-converted bf16 feature (only if workspace is big enough).
  __bf16* feat_bf   = (__bf16*)carve((size_t)M * E * 2);
  const bool pre_convert_feature = (off <= ws_size);

  // 1) weight conversions fp32 -> bf16 (all sizes divisible by 4)
  cvt4_bf16_kernel<<<(H * E / 4 + 255) / 256, 256, 0, stream>>>(
      W_embed, (uint2*)wembed_bf, H * E / 4);
  cvt4_bf16_kernel<<<(3 * H * H / 4 + 255) / 256, 256, 0, stream>>>(
      W_ih, (uint2*)wih_bf, 3 * H * H / 4);
  cvt4_bf16_kernel<<<(3 * H * H / 4 + 255) / 256, 256, 0, stream>>>(
      W_hh, (uint2*)whh_bf, 3 * H * H / 4);

  // 2) embed: x = tanh(feature @ W_embed^T + b_embed)  -> bf16 [15360,1024]
  if (pre_convert_feature) {
    cvt4_bf16_kernel<<<((size_t)M * E / 4 + 255) / 256, 256, 0, stream>>>(
        feature, (uint2*)feat_bf, M * E / 4);
    wmma_gemm_kernel<1, false><<<dim3(H / 64, M / 128), 256, 0, stream>>>(
        feat_bf, wembed_bf, b_embed, x_bf, M, H, E);
  } else {
    wmma_gemm_kernel<1, true><<<dim3(H / 64, M / 128), 256, 0, stream>>>(
        feature, wembed_bf, b_embed, x_bf, M, H, E);
  }

  // 3) gx = x @ W_ih^T + b_ih  -> fp32 [15360,3072]
  wmma_gemm_kernel<0, false><<<dim3(3 * H / 64, M / 128), 256, 0, stream>>>(
      x_bf, wih_bf, b_ih, gx, M, 3 * H, H);

  // 4) GRU recurrence
  gru_init_kernel<<<(MR * H + 255) / 256, 256, 0, stream>>>(h32, hbf, MR * H);
  for (int t = 0; t < T; ++t) {
    wmma_gemm_kernel<0, false><<<dim3(3 * H / 64, MR / 128), 256, 0, stream>>>(
        hbf, whh_bf, b_hh, gh, MR, 3 * H, H);
    gru_pointwise_kernel<<<(MR * H) / 256, 256, 0, stream>>>(
        gx, gh, h32, hbf, hiddens, t);
  }

  // 5) heads
  float* out_action   = (float*)d_out;               // [15360, 9]
  float* out_activity = out_action + (size_t)M * 9;  // [1280, 8]
  action_head_kernel<<<M, 32, 0, stream>>>(hiddens, W_act, b_act, out_action);
  activity_head_kernel<<<B * T, 32, 0, stream>>>(hiddens, W_activ, b_activ, out_activity);
}
